// UCR_44762149159465
// MI455X (gfx1250) — matrix operand order repaired and checked by reading, not compiled, not run on hardware
//
#include <hip/hip_runtime.h>
#include <hip/hip_bf16.h>

// Problem constants (match reference)
constexpr int NROWS = 8192;
constexpr int DDIM  = 1024;
constexpr float INV_TEMP = 20.0f;   // 1/0.05
constexpr float SHIFT    = 20.0f;   // max possible sim/temp, fixed LSE shift
constexpr float EPSN     = 1e-8f;

typedef _Float16 v16h __attribute__((ext_vector_type(16)));
typedef _Float16 v8h  __attribute__((ext_vector_type(8)));
typedef float    v8f  __attribute__((ext_vector_type(8)));

union H16 { v16h v; v8h h[2]; };

// ---------------------------------------------------------------------------
// Kernel 1: per-row norms, exact fp32 diagonal, fp16 conversion.
//   xnS = (x/||x||) * (1/TEMP)  (scale folded into A operand)
//   ynh = (y/||y||)
//   diag[i] = dot(xn_i, yn_i) / TEMP   (computed in fp32, not via WMMA)
// ---------------------------------------------------------------------------
__global__ void __launch_bounds__(256)
prep_kernel(const float* __restrict__ x, const float* __restrict__ m,
            _Float16* __restrict__ xnS, _Float16* __restrict__ ynh,
            float* __restrict__ diag) {
  const int row = blockIdx.x;
  const int t   = threadIdx.x;
  const float* xr = x + (size_t)row * DDIM;
  const float* mr = m + (size_t)row * DDIM;

  float xv[4], yv[4];
  float sxx = 0.f, syy = 0.f, sxy = 0.f;
#pragma unroll
  for (int i = 0; i < 4; ++i) {
    const int idx = t + i * 256;
    const float xx = xr[idx];
    const float yy = xx * mr[idx];
    xv[i] = xx; yv[i] = yy;
    sxx += xx * xx; syy += yy * yy; sxy += xx * yy;
  }
#pragma unroll
  for (int o = 16; o >= 1; o >>= 1) {
    sxx += __shfl_xor(sxx, o, 32);
    syy += __shfl_xor(syy, o, 32);
    sxy += __shfl_xor(sxy, o, 32);
  }
  __shared__ float red[3][8];
  const int wave = t >> 5, lane = t & 31;
  if (lane == 0) { red[0][wave] = sxx; red[1][wave] = syy; red[2][wave] = sxy; }
  __syncthreads();
  float txx = 0.f, tyy = 0.f, txy = 0.f;
#pragma unroll
  for (int w = 0; w < 8; ++w) { txx += red[0][w]; tyy += red[1][w]; txy += red[2][w]; }

  const float invx = 1.0f / fmaxf(sqrtf(txx), EPSN);
  const float invy = 1.0f / fmaxf(sqrtf(tyy), EPSN);
  if (t == 0) diag[row] = txy * invx * invy * INV_TEMP;

  const float sx = invx * INV_TEMP;
#pragma unroll
  for (int i = 0; i < 4; ++i) {
    const int idx = t + i * 256;
    xnS[(size_t)row * DDIM + idx] = (_Float16)(xv[i] * sx);
    ynh[(size_t)row * DDIM + idx] = (_Float16)(yv[i] * invy);
  }
}

__global__ void zero_kernel(float* __restrict__ p, int n) {
  const int i = blockIdx.x * blockDim.x + threadIdx.x;
  if (i < n) p[i] = 0.f;
}

// ---------------------------------------------------------------------------
// Kernel 2: fused GEMM + shifted exp row-sum.
// WG = 8 waves, 128-row block, grid.y = 8 column splits of 1024 cols.
// Waves arranged 2 row-groups x 4 col-groups; each wave computes
// 4 row-tiles x 4 col-tiles (64x64) so each LDS A-fragment feeds 4 WMMAs
// (~256 B LDS traffic per WMMA, matching LDS bandwidth).
// A chunks staged via async global->LDS (ASYNCcnt), B streamed from L2.
// ---------------------------------------------------------------------------
constexpr int BM = 128;   // rows per workgroup
constexpr int BN = 256;   // columns per j-block (4 col-groups * 64)
constexpr int KC = 64;    // K chunk staged in LDS (2 WMMA K-steps)
constexpr int COLSPLIT = 8;
constexpr int COLSPAN  = NROWS / COLSPLIT;  // 1024 columns per WG

__global__ void __launch_bounds__(256)
gemm_lse_kernel(const _Float16* __restrict__ xnS,
                const _Float16* __restrict__ ynh,
                float* __restrict__ rowsum) {
  __shared__ __align__(16) _Float16 As[BM][KC + 8];  // stride 72 halfs = 144B

  const int row0 = blockIdx.x * BM;
  const int col0 = blockIdx.y * COLSPAN;
  const int tid  = threadIdx.x;
  const int wave = tid >> 5;
  const int lane = tid & 31;
  const int l16  = lane & 15;
  const int hi   = lane >> 4;
  const int rowgrp = wave >> 2;          // 0..1: rows rowgrp*64 .. +64
  const int colgrp = wave & 3;           // 0..3: cols colgrp*64 .. +64
  const int rowbase = rowgrp * 64;

  const v8f vzero = {0.f, 0.f, 0.f, 0.f, 0.f, 0.f, 0.f, 0.f};
  v8f racc[4];
#pragma unroll
  for (int t = 0; t < 4; ++t) racc[t] = vzero;

  for (int jb = 0; jb < COLSPAN; jb += BN) {
    const int colw = col0 + jb + colgrp * 64;   // this wave's 64 columns
    v8f C[16];
#pragma unroll
    for (int t = 0; t < 16; ++t) C[t] = vzero;

    for (int kc = 0; kc < DDIM; kc += KC) {
      __syncthreads();
      {
        // Stage A chunk: 128 rows x 64 halfs; 2 threads/row, 64B each,
        // via CDNA5 async global->LDS (same IOFFSET applied to both sides).
        const int r = tid >> 1, seg = tid & 1;
        const unsigned lds =
            (unsigned)(uintptr_t)(&As[r][seg * 32]);   // low 32b = LDS addr
        const _Float16* g = xnS + (size_t)(row0 + r) * DDIM + kc + seg * 32;
        asm volatile(
            "global_load_async_to_lds_b128 %0, %1, off\n\t"
            "global_load_async_to_lds_b128 %0, %1, off offset:16\n\t"
            "global_load_async_to_lds_b128 %0, %1, off offset:32\n\t"
            "global_load_async_to_lds_b128 %0, %1, off offset:48"
            :: "v"(lds), "v"(g) : "memory");
        asm volatile("s_wait_asynccnt 0x0" ::: "memory");
      }
      __syncthreads();
#pragma unroll
      for (int ks = 0; ks < KC; ks += 32) {
        // B fragments: lane l16 = column; K 0..15 (lo lanes) / 16..31 (hi
        // lanes) contiguous in row-major ynh.
        v16h b[4];
#pragma unroll
        for (int n = 0; n < 4; ++n) {
          const _Float16* bp =
              ynh + (size_t)(colw + n * 16 + l16) * DDIM + kc + ks + hi * 16;
          b[n] = *(const v16h*)bp;
        }
#pragma unroll
        for (int t = 0; t < 4; ++t) {
          // A fragment: lane l16 = row; lo lanes K 0..15, hi lanes
          // K 16..31 (contiguous, matching B). Reused 4x.
          H16 a;
          const _Float16* ap = &As[rowbase + t * 16 + l16][ks + hi * 16];
          a.h[0] = *(const v8h*)ap;
          a.h[1] = *(const v8h*)(ap + 8);
#pragma unroll
          for (int n = 0; n < 4; ++n)
            C[t * 4 + n] = __builtin_amdgcn_wmma_f32_16x16x32_f16(
                false, a.v, false, b[n], (short)0, C[t * 4 + n], false, false);
        }
      }
    }
    // Shifted exp accumulate (exp(s-20) <= 1 since cos<=1 -> no online max).
#pragma unroll
    for (int t = 0; t < 4; ++t)
#pragma unroll
      for (int n = 0; n < 4; ++n)
#pragma unroll
        for (int r = 0; r < 8; ++r)
          racc[t][r] += __expf(C[t * 4 + n][r] - SHIFT);
  }

  // Reduce each row's 16 column-partials (xor 1,2,4,8 stays inside the
  // 16-lane half that owns the row), then one atomic per row per wave.
#pragma unroll
  for (int t = 0; t < 4; ++t)
#pragma unroll
    for (int r = 0; r < 8; ++r) {
      float v = racc[t][r];
      v += __shfl_xor(v, 1, 32);
      v += __shfl_xor(v, 2, 32);
      v += __shfl_xor(v, 4, 32);
      v += __shfl_xor(v, 8, 32);
      if (l16 == 0) {
        const int row = row0 + rowbase + t * 16 + hi * 8 + r;
        atomicAdd(&rowsum[row], v);
      }
    }
}

// ---------------------------------------------------------------------------
// Kernel 3: loss = mean_i (SHIFT + log(rowsum_i) - diag_i)
// ---------------------------------------------------------------------------
__global__ void __launch_bounds__(256)
finalize_kernel(const float* __restrict__ rowsum,
                const float* __restrict__ diag, float* __restrict__ out) {
  const int t = threadIdx.x;
  float s = 0.f;
  for (int i = t; i < NROWS; i += 256)
    s += SHIFT + __logf(rowsum[i]) - diag[i];
#pragma unroll
  for (int o = 16; o >= 1; o >>= 1) s += __shfl_xor(s, o, 32);
  __shared__ float red[8];
  if ((t & 31) == 0) red[t >> 5] = s;
  __syncthreads();
  if (t == 0) {
    float tot = 0.f;
#pragma unroll
    for (int w = 0; w < 8; ++w) tot += red[w];
    out[0] = tot / (float)NROWS;
  }
}

// ---------------------------------------------------------------------------
extern "C" void kernel_launch(void* const* d_in, const int* in_sizes, int n_in,
                              void* d_out, int out_size, void* d_ws, size_t ws_size,
                              hipStream_t stream) {
  (void)in_sizes; (void)n_in; (void)out_size; (void)ws_size;
  const float* x = (const float*)d_in[0];
  const float* m = (const float*)d_in[1];
  float* out = (float*)d_out;

  // Workspace layout: xnS (16MB f16) | ynh (16MB f16) | diag (32KB) | rowsum (32KB)
  char* w = (char*)d_ws;
  _Float16* xnS = (_Float16*)w;
  _Float16* ynh = xnS + (size_t)NROWS * DDIM;
  float* diag   = (float*)(ynh + (size_t)NROWS * DDIM);
  float* rowsum = diag + NROWS;

  prep_kernel<<<NROWS, 256, 0, stream>>>(x, m, xnS, ynh, diag);
  zero_kernel<<<NROWS / 256, 256, 0, stream>>>(rowsum, NROWS);
  gemm_lse_kernel<<<dim3(NROWS / BM, COLSPLIT), 256, 0, stream>>>(xnS, ynh, rowsum);
  finalize_kernel<<<1, 256, 0, stream>>>(rowsum, diag, out);
}
